// SelfAttention_52407190945982
// MI455X (gfx1250) — matrix operand order repaired and checked by reading, not compiled, not run on hardware
//
#include <hip/hip_runtime.h>
#include <stdint.h>

// ---------------- problem constants ----------------
#define NB   4
#define NT   2048
#define NE   1024
#define NH   16
#define ND   64
#define NM   (NB*NT)        // 8192 rows

#define USE_ASYNC 1

typedef __attribute__((ext_vector_type(16))) __bf16 v16bf;
typedef __attribute__((ext_vector_type(8)))  float  v8f;

__device__ __forceinline__ uint16_t f2bf(float x) {
  uint32_t u = __builtin_bit_cast(uint32_t, x);
  uint32_t r = (u + 0x7FFFu + ((u >> 16) & 1u)) >> 16;
  return (uint16_t)r;
}

__device__ __forceinline__ v16bf ld_frag(const uint16_t* p0, const uint16_t* p1) {
  union { v16bf v; uint4 q[2]; } f;
  f.q[0] = *reinterpret_cast<const uint4*>(p0);
  f.q[1] = *reinterpret_cast<const uint4*>(p1);
  return f.v;
}

__device__ __forceinline__ v8f wmma_bf16(v16bf a, v16bf b, v8f c) {
  return __builtin_amdgcn_wmma_f32_16x16x32_bf16(false, a, false, b, (short)0, c, false, false);
}

// ---- 32-byte memory -> LDS copy (async on CDNA5, tracked by ASYNCcnt) ----
__device__ __forceinline__ void cp32_to_lds(uint16_t* lds, const uint16_t* g) {
#if USE_ASYNC
  uint32_t l = (uint32_t)(uintptr_t)lds;          // LDS addr = generic addr[31:0]
  uint64_t a = (uint64_t)(uintptr_t)g;
  asm volatile(
      "global_load_async_to_lds_b128 %0, %1, off\n\t"
      "global_load_async_to_lds_b128 %0, %1, off offset:16"
      :: "v"(l), "v"(a) : "memory");
#else
  const uint4* s = reinterpret_cast<const uint4*>(g);
  uint4 a0 = s[0], a1 = s[1];
  uint4* d = reinterpret_cast<uint4*>(lds);
  d[0] = a0; d[1] = a1;
#endif
}

__device__ __forceinline__ void wait_async() {
#if USE_ASYNC
  asm volatile("s_wait_asynccnt 0x0" ::: "memory");
#endif
}

// ---------------- fp32 -> bf16 convert ----------------
__global__ void cvt_bf16_kernel(const float* __restrict__ src,
                                uint16_t* __restrict__ dst, long n) {
  long i = (long)blockIdx.x * blockDim.x + threadIdx.x;
  long stride = (long)gridDim.x * blockDim.x;
  for (; i < n; i += stride) dst[i] = f2bf(src[i]);
}

// fp32 [K][N] -> bf16 transposed [N][K]
__global__ void cvt_bf16_t_kernel(const float* __restrict__ src,
                                  uint16_t* __restrict__ dst, int K, int N) {
  long total = (long)K * N;
  long i = (long)blockIdx.x * blockDim.x + threadIdx.x;
  long stride = (long)gridDim.x * blockDim.x;
  for (; i < total; i += stride) {
    int n = (int)(i / K), k = (int)(i % K);
    dst[i] = f2bf(src[(size_t)k * N + n]);
  }
}

// ---------------- QKV GEMM: xb[8192,1024] @ wT[3072,1024]^T + bias ----------------
// writes Q,K bf16 in [b,h,t,d]; V bf16 transposed in [b,h,d,t]
__global__ __launch_bounds__(256)
void gemm_qkv_kernel(const uint16_t* __restrict__ xb,
                     const uint16_t* __restrict__ wT,   // [3072][1024]
                     const float* __restrict__ bias,
                     uint16_t* __restrict__ qout,
                     uint16_t* __restrict__ kout,
                     uint16_t* __restrict__ vout) {
  __shared__ __align__(16) uint16_t As[2][128 * 40];   // [row][k] pad 8
  __shared__ __align__(16) uint16_t Bs[2][128 * 40];   // [n][k]  pad 8
  const int tid  = threadIdx.x;
  const int lane = tid & 31, w = tid >> 5;
  const int wr = w >> 1, wc = w & 1;        // wave grid 4x2: 32 rows x 64 cols
  const int lh = lane >> 4, lm = lane & 15;
  const int row0 = blockIdx.y * 128;
  const int col0 = blockIdx.x * 128;

  const int sr = tid >> 1, sh = tid & 1;    // staging: 2 threads per 32-elem row

  v8f acc[2][4];
#pragma unroll
  for (int mt = 0; mt < 2; ++mt)
#pragma unroll
    for (int nt = 0; nt < 4; ++nt)
      acc[mt][nt] = (v8f){0.f,0.f,0.f,0.f,0.f,0.f,0.f,0.f};

  auto issue = [&](int buf, int kk) {
    cp32_to_lds(&As[buf][sr * 40 + sh * 16],
                xb + (size_t)(row0 + sr) * NE + kk + sh * 16);
    cp32_to_lds(&Bs[buf][sr * 40 + sh * 16],
                wT + (size_t)(col0 + sr) * NE + kk + sh * 16);
  };

  issue(0, 0);
  const int KT = NE / 32;                   // 32 k-tiles
  for (int ki = 0; ki < KT; ++ki) {
    wait_async();
    __syncthreads();
    if (ki + 1 < KT) issue((ki + 1) & 1, (ki + 1) * 32);
    const uint16_t* Ab = As[ki & 1];
    const uint16_t* Bb = Bs[ki & 1];

    v16bf af[2], bfr[4];
#pragma unroll
    for (int mt = 0; mt < 2; ++mt) {
      const uint16_t* p = Ab + (wr * 32 + mt * 16 + lm) * 40 + lh * 8;
      af[mt] = ld_frag(p, p + 16);
    }
#pragma unroll
    for (int nt = 0; nt < 4; ++nt) {
      const uint16_t* p = Bb + (wc * 64 + nt * 16 + lm) * 40 + lh * 8;
      bfr[nt] = ld_frag(p, p + 16);
    }
#pragma unroll
    for (int mt = 0; mt < 2; ++mt)
#pragma unroll
      for (int nt = 0; nt < 4; ++nt)
        acc[mt][nt] = wmma_bf16(af[mt], bfr[nt], acc[mt][nt]);
    __syncthreads();
  }

  // ---- epilogue: scatter to Q/K [b,h,t,d] and V [b,h,d,t], add bias ----
#pragma unroll
  for (int mt = 0; mt < 2; ++mt) {
#pragma unroll
    for (int nt = 0; nt < 4; ++nt) {
#pragma unroll
      for (int v = 0; v < 8; ++v) {
        int m = wr * 32 + mt * 16 + v + lh * 8;
        int r = row0 + m;
        int c = col0 + wc * 64 + nt * 16 + lm;
        float val = acc[mt][nt][v] + bias[c];
        int which = c >> 10;     // 0=q 1=k 2=v
        int e = c & 1023;
        int h = e >> 6, d = e & 63;
        int b = r >> 11, t = r & 2047;
        size_t bh = (size_t)(b * NH + h);
        uint16_t bv = f2bf(val);
        if (which == 0)      qout[(bh * NT + t) * ND + d] = bv;
        else if (which == 1) kout[(bh * NT + t) * ND + d] = bv;
        else                 vout[(bh * ND + d) * NT + t] = bv;   // transposed
      }
    }
  }
}

// ---------------- flash attention ----------------
// grid: (T/64, B*H); 128 threads = 4 waves, each wave handles 16 query rows
__global__ __launch_bounds__(128)
void attn_kernel(const uint16_t* __restrict__ Qb,    // [bh][t][d]
                 const uint16_t* __restrict__ Kb,    // [bh][t][d]
                 const uint16_t* __restrict__ Vt,    // [bh][d][t]
                 uint16_t* __restrict__ yb) {
  __shared__ __align__(16) uint16_t Ks[2][32 * 72];   // [kv][d] pad 8
  __shared__ __align__(16) uint16_t Vs[2][64 * 40];   // [d][kv] pad 8
  __shared__ __align__(16) uint16_t Ps[4][16 * 40];   // per-wave P tile [m][kv]

  const int tid = threadIdx.x, lane = tid & 31, w = tid >> 5;
  const int lh = lane >> 4, lm = lane & 15;
  const int bh = blockIdx.y;
  const int b = bh >> 4, h = bh & 15;
  const int q0 = blockIdx.x * 64;
  const uint16_t* Qh = Qb + (size_t)bh * NT * ND;
  const uint16_t* Kh = Kb + (size_t)bh * NT * ND;
  const uint16_t* Vh = Vt + (size_t)bh * ND * NT;

  // staging maps
  const int kr = tid >> 2, ksub = tid & 3;   // K: 32 rows x 4 chunks of 16
  const int vr = tid >> 1, vsub = tid & 1;   // V: 64 rows x 2 chunks of 16

  auto issueKV = [&](int buf, int kv0) {
    cp32_to_lds(&Ks[buf][kr * 72 + ksub * 16],
                Kh + (size_t)(kv0 + kr) * ND + ksub * 16);
    cp32_to_lds(&Vs[buf][vr * 40 + vsub * 16],
                Vh + (size_t)vr * NT + kv0 + vsub * 16);
  };

  // Q fragments (A-layout, 2 k-steps over d=64), loaded once
  const int qrow = q0 + w * 16 + lm;
  v16bf qf[2];
#pragma unroll
  for (int ks = 0; ks < 2; ++ks) {
    const uint16_t* p = Qh + (size_t)qrow * ND + ks * 32 + lh * 8;
    qf[ks] = ld_frag(p, p + 16);
  }

  v8f o[4];
#pragma unroll
  for (int dt = 0; dt < 4; ++dt) o[dt] = (v8f){0.f,0.f,0.f,0.f,0.f,0.f,0.f,0.f};
  float mr[8], lr[8];
#pragma unroll
  for (int v = 0; v < 8; ++v) { mr[v] = -1e30f; lr[v] = 0.f; }

  const float scale = 0.125f;  // 1/sqrt(64)

  issueKV(0, 0);
  for (int kv0 = 0; kv0 < NT; kv0 += 32) {
    wait_async();
    __syncthreads();
    if (kv0 + 32 < NT) issueKV(((kv0 >> 5) + 1) & 1, kv0 + 32);
    const uint16_t* Kbuf = Ks[(kv0 >> 5) & 1];
    const uint16_t* Vbuf = Vs[(kv0 >> 5) & 1];

    // ---- S = Q . K^T  (16 x 32 tile) ----
    v8f s0 = (v8f){0.f,0.f,0.f,0.f,0.f,0.f,0.f,0.f};
    v8f s1 = (v8f){0.f,0.f,0.f,0.f,0.f,0.f,0.f,0.f};
#pragma unroll
    for (int ks = 0; ks < 2; ++ks) {
      const uint16_t* p0 = Kbuf + lm * 72 + ks * 32 + lh * 8;
      const uint16_t* p1 = Kbuf + (16 + lm) * 72 + ks * 32 + lh * 8;
      v16bf bk0 = ld_frag(p0, p0 + 16);
      v16bf bk1 = ld_frag(p1, p1 + 16);
      s0 = wmma_bf16(qf[ks], bk0, s0);
      s1 = wmma_bf16(qf[ks], bk1, s1);
    }

    // ---- online softmax (row m = v + 8*lanehalf; cols across 16 lanes) ----
    float p0v[8], p1v[8];
#pragma unroll
    for (int v = 0; v < 8; ++v) {
      float a = s0[v] * scale;
      float c = s1[v] * scale;
      float rm = fmaxf(a, c);
      rm = fmaxf(rm, __shfl_xor(rm, 1, 32));
      rm = fmaxf(rm, __shfl_xor(rm, 2, 32));
      rm = fmaxf(rm, __shfl_xor(rm, 4, 32));
      rm = fmaxf(rm, __shfl_xor(rm, 8, 32));
      float mnew = fmaxf(mr[v], rm);
      float corr = __expf(mr[v] - mnew);
      float p0 = __expf(a - mnew);
      float p1 = __expf(c - mnew);
      float rs = p0 + p1;
      rs += __shfl_xor(rs, 1, 32);
      rs += __shfl_xor(rs, 2, 32);
      rs += __shfl_xor(rs, 4, 32);
      rs += __shfl_xor(rs, 8, 32);
      lr[v] = lr[v] * corr + rs;
      mr[v] = mnew;
      o[0][v] *= corr; o[1][v] *= corr; o[2][v] *= corr; o[3][v] *= corr;
      p0v[v] = p0; p1v[v] = p1;
    }

    // ---- route P through LDS: C-layout -> A-layout (per-wave, DS in-order) ----
    uint16_t* P = &Ps[w][0];
#pragma unroll
    for (int v = 0; v < 8; ++v) {
      int m = v + lh * 8;
      P[m * 40 + lm]      = f2bf(p0v[v]);
      P[m * 40 + 16 + lm] = f2bf(p1v[v]);
    }
    asm volatile("s_wait_dscnt 0x0" ::: "memory");
    const uint16_t* pp = P + lm * 40 + lh * 8;
    v16bf pf = ld_frag(pp, pp + 16);

    // ---- O += P . V ----
#pragma unroll
    for (int dt = 0; dt < 4; ++dt) {
      const uint16_t* vp = Vbuf + (dt * 16 + lm) * 40 + lh * 8;
      v16bf vf = ld_frag(vp, vp + 16);
      o[dt] = wmma_bf16(pf, vf, o[dt]);
    }
    __syncthreads();
  }

  // ---- normalize and write y (bf16, [b,t, h*64+d]) ----
  float inv[8];
#pragma unroll
  for (int v = 0; v < 8; ++v) inv[v] = 1.f / lr[v];
#pragma unroll
  for (int dt = 0; dt < 4; ++dt) {
#pragma unroll
    for (int v = 0; v < 8; ++v) {
      int m = v + lh * 8;
      int row = q0 + w * 16 + m;
      int col = h * ND + dt * 16 + lm;
      yb[((size_t)b * NT + row) * NE + col] = f2bf(o[dt][v] * inv[v]);
    }
  }
}

// ---------------- output GEMM: yb[8192,1024] @ woT[1024,1024]^T + out_b -> fp32 ----------------
__global__ __launch_bounds__(256)
void gemm_out_kernel(const uint16_t* __restrict__ yb,
                     const uint16_t* __restrict__ wT,   // [1024][1024] transposed
                     const float* __restrict__ bias,
                     float* __restrict__ out) {
  __shared__ __align__(16) uint16_t As[2][128 * 40];
  __shared__ __align__(16) uint16_t Bs[2][128 * 40];
  const int tid  = threadIdx.x;
  const int lane = tid & 31, w = tid >> 5;
  const int wr = w >> 1, wc = w & 1;
  const int lh = lane >> 4, lm = lane & 15;
  const int row0 = blockIdx.y * 128;
  const int col0 = blockIdx.x * 128;
  const int sr = tid >> 1, sh = tid & 1;

  v8f acc[2][4];
#pragma unroll
  for (int mt = 0; mt < 2; ++mt)
#pragma unroll
    for (int nt = 0; nt < 4; ++nt)
      acc[mt][nt] = (v8f){0.f,0.f,0.f,0.f,0.f,0.f,0.f,0.f};

  auto issue = [&](int buf, int kk) {
    cp32_to_lds(&As[buf][sr * 40 + sh * 16],
                yb + (size_t)(row0 + sr) * NE + kk + sh * 16);
    cp32_to_lds(&Bs[buf][sr * 40 + sh * 16],
                wT + (size_t)(col0 + sr) * NE + kk + sh * 16);
  };

  issue(0, 0);
  const int KT = NE / 32;
  for (int ki = 0; ki < KT; ++ki) {
    wait_async();
    __syncthreads();
    if (ki + 1 < KT) issue((ki + 1) & 1, (ki + 1) * 32);
    const uint16_t* Ab = As[ki & 1];
    const uint16_t* Bb = Bs[ki & 1];

    v16bf af[2], bfr[4];
#pragma unroll
    for (int mt = 0; mt < 2; ++mt) {
      const uint16_t* p = Ab + (wr * 32 + mt * 16 + lm) * 40 + lh * 8;
      af[mt] = ld_frag(p, p + 16);
    }
#pragma unroll
    for (int nt = 0; nt < 4; ++nt) {
      const uint16_t* p = Bb + (wc * 64 + nt * 16 + lm) * 40 + lh * 8;
      bfr[nt] = ld_frag(p, p + 16);
    }
#pragma unroll
    for (int mt = 0; mt < 2; ++mt)
#pragma unroll
      for (int nt = 0; nt < 4; ++nt)
        acc[mt][nt] = wmma_bf16(af[mt], bfr[nt], acc[mt][nt]);
    __syncthreads();
  }

#pragma unroll
  for (int mt = 0; mt < 2; ++mt) {
#pragma unroll
    for (int nt = 0; nt < 4; ++nt) {
#pragma unroll
      for (int v = 0; v < 8; ++v) {
        int m = wr * 32 + mt * 16 + v + lh * 8;
        int r = row0 + m;
        int c = col0 + wc * 64 + nt * 16 + lm;
        out[(size_t)r * NE + c] = acc[mt][nt][v] + bias[c];
      }
    }
  }
}

// ---------------- host launch ----------------
extern "C" void kernel_launch(void* const* d_in, const int* in_sizes, int n_in,
                              void* d_out, int out_size, void* d_ws, size_t ws_size,
                              hipStream_t stream) {
  const float* x     = (const float*)d_in[0];
  const float* qkv_w = (const float*)d_in[1];
  const float* qkv_b = (const float*)d_in[2];
  const float* out_w = (const float*)d_in[3];
  const float* out_b = (const float*)d_in[4];
  float* out = (float*)d_out;

  char* ws = (char*)d_ws;
  size_t off = 0;
  auto alloc = [&](size_t bytes) -> void* {
    off = (off + 255) & ~(size_t)255;
    void* p = ws + off;
    off += bytes;
    return p;
  };
  const size_t nx  = (size_t)NM * NE;        // 8,388,608
  const size_t nwq = (size_t)NE * 3 * NE;    // 3,145,728
  const size_t nwo = (size_t)NE * NE;        // 1,048,576
  uint16_t* xb  = (uint16_t*)alloc(nx  * 2);
  uint16_t* wqT = (uint16_t*)alloc(nwq * 2);  // [3072][1024]
  uint16_t* woT = (uint16_t*)alloc(nwo * 2);  // [1024][1024]
  uint16_t* Qb  = (uint16_t*)alloc(nx * 2);   // [bh][t][d]
  uint16_t* Kb  = (uint16_t*)alloc(nx * 2);   // [bh][t][d]
  uint16_t* Vt  = (uint16_t*)alloc(nx * 2);   // [bh][d][t]
  uint16_t* yb  = (uint16_t*)alloc(nx * 2);

  // 1) convert (and transpose weights) to bf16
  cvt_bf16_kernel  <<<4096, 256, 0, stream>>>(x, xb, (long)nx);
  cvt_bf16_t_kernel<<<4096, 256, 0, stream>>>(qkv_w, wqT, NE, 3 * NE);
  cvt_bf16_t_kernel<<<2048, 256, 0, stream>>>(out_w, woT, NE, NE);

  // 2) QKV projection
  dim3 g1(3 * NE / 128, NM / 128);           // (24, 64)
  gemm_qkv_kernel<<<g1, 256, 0, stream>>>(xb, wqT, qkv_b, Qb, Kb, Vt);

  // 3) flash attention
  dim3 g2(NT / 64, NB * NH);                 // (32, 64)
  attn_kernel<<<g2, 128, 0, stream>>>(Qb, Kb, Vt, yb);

  // 4) output projection
  dim3 g3(NE / 128, NM / 128);               // (8, 64)
  gemm_out_kernel<<<g3, 256, 0, stream>>>(yb, woT, out_b, out);
}
